// MultiHeadGraphAttentionLayer_2774548873387
// MI455X (gfx1250) — compile-verified, matrix operand order checked by reading, not run on hardware
//
#include <hip/hip_runtime.h>
#include <math.h>

// ---------------- problem constants (from reference) ----------------
#define N_NODES 4096
#define IN_F    512
#define OUT_F   64
#define HEADS   8
#define ALPHA   0.2f
#define NEG_BIG (-9000000000000000.0f)

typedef __attribute__((ext_vector_type(16))) __bf16 v16bf;
typedef __attribute__((ext_vector_type(8)))  float  v8f;

union AFrag { v16bf v; __bf16 h[16]; unsigned short s[16]; uint4 q[2]; };
union CFrag { v8f   v; float f[8]; };

__device__ __forceinline__ unsigned short f2bf(float f) {
    union { float f; unsigned int u; } v; v.f = f;
    unsigned int r = v.u + 0x7FFFu + ((v.u >> 16) & 1u);   // round-nearest-even
    return (unsigned short)(r >> 16);
}
__device__ __forceinline__ float bf2f(unsigned short s) {
    union { float f; unsigned int u; } v; v.u = ((unsigned int)s) << 16;
    return v.f;
}

// ---------------- conversion kernels ----------------
__global__ void cvt_h_kernel(const float* __restrict__ h, unsigned short* __restrict__ hbf) {
    int i = blockIdx.x * blockDim.x + threadIdx.x;
    if (i < N_NODES * IN_F) hbf[i] = f2bf(h[i]);
}

// W[h][k][o] (f32) -> WtB[(h*OUT_F+o)][k] (bf16), K-contiguous B operand
__global__ void cvt_W_kernel(const float* __restrict__ W, unsigned short* __restrict__ WtB) {
    int i = blockIdx.x * blockDim.x + threadIdx.x;
    if (i >= HEADS * IN_F * OUT_F) return;
    int hd  = i / (IN_F * OUT_F);
    int rem = i - hd * (IN_F * OUT_F);
    int k   = rem / OUT_F;
    int o   = rem - k * OUT_F;
    WtB[(hd * OUT_F + o) * IN_F + k] = f2bf(W[i]);
}

// ---------------- GEMM1: Wh = h @ W  (per head), write WhT[o][n] bf16 ----------------
// grid: 256 blocks (n-tiles of 16), block: 256 threads = 8 waves = 8 heads
__global__ void gat_gemm1_kernel(const unsigned short* __restrict__ hbf,
                                 const unsigned short* __restrict__ WtB,
                                 unsigned short* __restrict__ WhT) {
    const int lane = threadIdx.x & 31;
    const int hd   = threadIdx.x >> 5;        // wave id == head
    const int n0   = blockIdx.x * 16;
    const int rsel = lane & 15;
    const int hi   = lane >> 4;               // lane-half selects K sub-block

    CFrag c[4];
#pragma unroll
    for (int t = 0; t < 4; ++t)
#pragma unroll
        for (int r = 0; r < 8; ++r) c[t].f[r] = 0.0f;

    const unsigned short* arow  = hbf + (size_t)(n0 + rsel) * IN_F;
    const unsigned short* bbase = WtB + (size_t)hd * OUT_F * IN_F;

    for (int kb = 0; kb < IN_F; kb += 32) {
        AFrag a;
        a.q[0] = *(const uint4*)(arow + kb +      hi * 8);   // K = kb+{0..7|8..15}
        a.q[1] = *(const uint4*)(arow + kb + 16 + hi * 8);   // K = kb+16+{...}
#pragma unroll
        for (int t = 0; t < 4; ++t) {
            const unsigned short* brow = bbase + (size_t)(t * 16 + rsel) * IN_F;
            AFrag b;
            b.q[0] = *(const uint4*)(brow + kb +      hi * 8);
            b.q[1] = *(const uint4*)(brow + kb + 16 + hi * 8);
            c[t].v = __builtin_amdgcn_wmma_f32_16x16x32_bf16(
                false, a.v, false, b.v, (short)0, c[t].v, false, false);
        }
    }

    // C layout: VGPR r -> (M = r + hi*8, N = rsel). Store transposed: WhT[(hd*64+o)][n]
#pragma unroll
    for (int t = 0; t < 4; ++t) {
        const int o = t * 16 + rsel;
#pragma unroll
        for (int r = 0; r < 8; ++r) {
            const int n = n0 + r + hi * 8;
            WhT[(size_t)(hd * OUT_F + o) * N_NODES + n] = f2bf(c[t].f[r]);
        }
    }
}

// ---------------- Wh1/Wh2 = Wh . a1 / a2 ----------------
__global__ void gat_wh12_kernel(const unsigned short* __restrict__ WhT,
                                const float* __restrict__ a,
                                float* __restrict__ Wh1, float* __restrict__ Wh2) {
    int i = blockIdx.x * blockDim.x + threadIdx.x;     // i = hd*4096 + n
    if (i >= HEADS * N_NODES) return;
    int hd = i >> 12;
    int n  = i & (N_NODES - 1);
    float s1 = 0.0f, s2 = 0.0f;
#pragma unroll 8
    for (int o = 0; o < OUT_F; ++o) {
        float w = bf2f(WhT[(size_t)(hd * OUT_F + o) * N_NODES + n]);
        s1 += w * a[hd * 2 * OUT_F + o];
        s2 += w * a[hd * 2 * OUT_F + OUT_F + o];
    }
    Wh1[i] = s1;
    Wh2[i] = s2;
}

// ---------------- GEMM2: fused attention build + (att @ Wh) + ELU ----------------
// grid: 256 blocks (n-tiles of 16), block: 256 threads = 8 waves = 8 heads.
// The 16x128 adjacency tile is copied into LDS once per block with
// global_load_async_to_lds_b128 (ASYNCcnt), double-buffered; all 8 heads
// consume it from LDS instead of issuing 8 redundant global streams.
#define STAGE_COLS   128
#define ROW_PAD_INTS 132                       // 128 + 4 pad -> conflict-free b128 reads
#define STAGE_LDS_B  (16 * ROW_PAD_INTS * 4)   // 8448 bytes per buffer

__global__ void gat_gemm2_kernel(const int* __restrict__ adj,
                                 const unsigned short* __restrict__ WhT,
                                 const float* __restrict__ Wh1,
                                 const float* __restrict__ Wh2,
                                 float* __restrict__ out) {
    __shared__ __attribute__((aligned(16))) int adjS[2][16][ROW_PAD_INTS];

    const int tid  = threadIdx.x;
    const int lane = tid & 31;
    const int hd   = tid >> 5;
    const int n0   = blockIdx.x * 16;
    const int rsel = lane & 15;
    const int hi   = lane >> 4;
    const int n    = n0 + rsel;

    // --- async-copy plumbing: each thread moves 2 x 16B per 16x128 stage ---
    const unsigned ldsBase = (unsigned)(unsigned long long)(void*)&adjS[0][0][0];
    const int f0   = tid * 16;                 // byte offset in unpadded 8192B stage
    const int row0 = f0 >> 9, colB0 = f0 & 511;
    const int f1   = f0 + 4096;
    const int row1 = f1 >> 9, colB1 = f1 & 511;
    const unsigned ldsOff0 = (unsigned)(row0 * (ROW_PAD_INTS * 4) + colB0);
    const unsigned ldsOff1 = (unsigned)(row1 * (ROW_PAD_INTS * 4) + colB1);
    const unsigned long long gRow0 =
        (unsigned long long)adj + (unsigned long long)(n0 + row0) * (N_NODES * 4) + colB0;
    const unsigned long long gRow1 =
        (unsigned long long)adj + (unsigned long long)(n0 + row1) * (N_NODES * 4) + colB1;

    #define ISSUE_STAGE(s)                                                          \
        do {                                                                        \
            unsigned _b  = (unsigned)((s) & 1) * STAGE_LDS_B;                       \
            unsigned long long _ga = (unsigned long long)(s) * (STAGE_COLS * 4);    \
            unsigned _d0 = ldsBase + _b + ldsOff0;                                  \
            unsigned _d1 = ldsBase + _b + ldsOff1;                                  \
            unsigned long long _a0 = gRow0 + _ga;                                   \
            unsigned long long _a1 = gRow1 + _ga;                                   \
            asm volatile("global_load_async_to_lds_b128 %0, %1, off"                \
                         :: "v"(_d0), "v"(_a0) : "memory");                         \
            asm volatile("global_load_async_to_lds_b128 %0, %1, off"                \
                         :: "v"(_d1), "v"(_a1) : "memory");                         \
        } while (0)

    CFrag c[4];
#pragma unroll
    for (int t = 0; t < 4; ++t)
#pragma unroll
        for (int r = 0; r < 8; ++r) c[t].f[r] = 0.0f;

    const float w1           = Wh1[hd * N_NODES + n];
    const float* wh2p        = Wh2 + (size_t)hd * N_NODES;
    const unsigned short* bb = WhT + (size_t)hd * OUT_F * N_NODES;

    const int S = N_NODES / STAGE_COLS;        // 32 stages
    ISSUE_STAGE(0);
    ISSUE_STAGE(1);

    for (int s = 0; s < S; ++s) {
        // my 2 copies for stage s are done when <= 2 (stage s+1) remain in flight
        if (s + 1 < S) asm volatile("s_wait_asynccnt 0x2" ::: "memory");
        else           asm volatile("s_wait_asynccnt 0x0" ::: "memory");
        __syncthreads();                        // everyone's stage-s copies landed

        const int buf = s & 1;
        const int m0  = s * STAGE_COLS;
#pragma unroll
        for (int kl = 0; kl < STAGE_COLS; kl += 32) {
            // ---- A fragment: attention tile built from LDS adj + Wh1/Wh2 ----
            int adjv[16];
            *(int4*)(adjv + 0)  = *(const int4*)&adjS[buf][rsel][kl +      hi * 8];
            *(int4*)(adjv + 4)  = *(const int4*)&adjS[buf][rsel][kl +  4 + hi * 8];
            *(int4*)(adjv + 8)  = *(const int4*)&adjS[buf][rsel][kl + 16 + hi * 8];
            *(int4*)(adjv + 12) = *(const int4*)&adjS[buf][rsel][kl + 20 + hi * 8];

            float w2v[16];
            const float* wp = wh2p + m0 + kl + hi * 8;
            *(float4*)(w2v + 0)  = *(const float4*)(wp);
            *(float4*)(w2v + 4)  = *(const float4*)(wp + 4);
            *(float4*)(w2v + 8)  = *(const float4*)(wp + 16);
            *(float4*)(w2v + 12) = *(const float4*)(wp + 20);

            AFrag af;
#pragma unroll
            for (int j = 0; j < 16; ++j) {
                float x = w1 + w2v[j];
                float e = fmaxf(x, ALPHA * x);               // leaky_relu as max
                af.h[j] = (__bf16)((adjv[j] > 0) ? e : NEG_BIG);
            }

            // ---- 4 B fragments (WhT, K-contiguous) + WMMAs ----
#pragma unroll
            for (int t = 0; t < 4; ++t) {
                const unsigned short* brow = bb + (size_t)(t * 16 + rsel) * N_NODES;
                AFrag b;
                b.q[0] = *(const uint4*)(brow + m0 + kl +      hi * 8);
                b.q[1] = *(const uint4*)(brow + m0 + kl + 16 + hi * 8);
                c[t].v = __builtin_amdgcn_wmma_f32_16x16x32_bf16(
                    false, af.v, false, b.v, (short)0, c[t].v, false, false);
            }
        }
        __syncthreads();                        // all waves done reading buf
        if (s + 2 < S) ISSUE_STAGE(s + 2);      // overwrite it for stage s+2
    }
    #undef ISSUE_STAGE

    // epilogue: ELU + scatter to out[n][hd*64+o]
#pragma unroll
    for (int t = 0; t < 4; ++t) {
        const int o = hd * OUT_F + t * 16 + rsel;
#pragma unroll
        for (int r = 0; r < 8; ++r) {
            const int nn = n0 + r + hi * 8;
            float x = c[t].f[r];
            out[(size_t)nn * (HEADS * OUT_F) + o] = (x > 0.0f) ? x : (__expf(x) - 1.0f);
        }
    }
}

// ---------------- host launcher ----------------
extern "C" void kernel_launch(void* const* d_in, const int* in_sizes, int n_in,
                              void* d_out, int out_size, void* d_ws, size_t ws_size,
                              hipStream_t stream) {
    const float* h   = (const float*)d_in[0];   // (4096, 512)  f32
    const int*   adj = (const int*)d_in[1];     // (4096, 4096) i32
    const float* W   = (const float*)d_in[2];   // (8, 512, 64) f32
    const float* a   = (const float*)d_in[3];   // (8, 128, 1)  f32
    float* out = (float*)d_out;                 // (4096, 512)  f32

    char* ws = (char*)d_ws;
    // workspace layout (total ~8.75 MB)
    unsigned short* hbf = (unsigned short*)(ws);                                        // 4 MB
    unsigned short* WtB = (unsigned short*)(ws + (size_t)4 * 1024 * 1024);              // 512 KB
    unsigned short* WhT = (unsigned short*)(ws + (size_t)4 * 1024 * 1024 + 512 * 1024); // 4 MB
    float* Wh1 = (float*)(ws + (size_t)8 * 1024 * 1024 + 512 * 1024);                   // 128 KB
    float* Wh2 = Wh1 + (size_t)HEADS * N_NODES;                                         // 128 KB

    cvt_h_kernel<<<(N_NODES * IN_F + 255) / 256, 256, 0, stream>>>(h, hbf);
    cvt_W_kernel<<<(HEADS * IN_F * OUT_F + 255) / 256, 256, 0, stream>>>(W, WtB);

    gat_gemm1_kernel<<<N_NODES / 16, 256, 0, stream>>>(hbf, WtB, WhT);

    gat_wh12_kernel<<<(HEADS * N_NODES + 255) / 256, 256, 0, stream>>>(WhT, a, Wh1, Wh2);

    gat_gemm2_kernel<<<N_NODES / 16, 256, 0, stream>>>(adj, WhT, Wh1, Wh2, out);
}
